// Q_PatchEmbed_63058709840192
// MI455X (gfx1250) — matrix operand order repaired and checked by reading, not compiled, not run on hardware
//
#include <hip/hip_runtime.h>
#include <hip/hip_bf16.h>

// Patch-embed as GEMM: C[25088,768] = A[25088,768] x B[768,768] + bias
//   A[m,k] = x[b, c, ph*16+py, pw*16+px],  m=(b,ph,pw), k=(c,py,px)
//   B[k,e] = W[e,k]  (W is [768][768] row-major in (e,k) already)
// f32 -> f16 convert on the fly, f32 accumulate via v_wmma_f32_16x16x32_f16.
// Double-buffered LDS, 1 barrier per K-step, global prefetch 1 iter ahead,
// running-pointer K addressing. Fragment loads batched into distinct regs
// (one dscnt wait per K-step), ds_stores issued after frag loads so WMMAs
// never wait on them.

typedef __attribute__((ext_vector_type(16))) _Float16 v16h;
typedef __attribute__((ext_vector_type(8)))  _Float16 v8h;
typedef __attribute__((ext_vector_type(8)))  float    v8f;
typedef __attribute__((ext_vector_type(4)))  float    v4f;

#define TM       128          // M tile per workgroup
#define TN       128          // N tile per workgroup
#define KSTEP    32           // K per WMMA step
#define LDA      40           // padded LDS row stride in halfs (80B = 20 banks)
#define NTHREADS 256          // 8 waves (wave32)

#define GEMM_K   768
#define GEMM_N   768
#define NPATCH   196          // 14*14 patches per image
#define NK       (GEMM_K / KSTEP)   // 24 (8 K-steps per input channel)

#define A_STEP_IN   (2 * 224)     // +2 image rows per K-step within a channel
#define A_STEP_CH   (210 * 224)   // row 14 of channel c -> row 0 of channel c+1

__device__ __forceinline__ v8h cvt8(v4f lo, v4f hi) {
    v8h r;
    r[0] = (_Float16)lo[0]; r[1] = (_Float16)lo[1];
    r[2] = (_Float16)lo[2]; r[3] = (_Float16)lo[3];
    r[4] = (_Float16)hi[0]; r[5] = (_Float16)hi[1];
    r[6] = (_Float16)hi[2]; r[7] = (_Float16)hi[3];
    return r;
}

__device__ __forceinline__ v16h ld16(const _Float16* p0, const _Float16* p1) {
    v8h a = *(const v8h*)p0;
    v8h b = *(const v8h*)p1;
    return __builtin_shufflevector(a, b,
               0,1,2,3,4,5,6,7,8,9,10,11,12,13,14,15);
}

__global__ __launch_bounds__(NTHREADS)
void patch_embed_wmma(const float* __restrict__ x,
                      const float* __restrict__ w,
                      const float* __restrict__ bias,
                      float* __restrict__ out)
{
    __shared__ _Float16 As[2][TM * LDA];   // 2 x 10 KB
    __shared__ _Float16 Bs[2][TN * LDA];   // 2 x 10 KB

    const int tid  = threadIdx.x;
    const int lane = tid & 31;
    const int wave = tid >> 5;
    const int wm   = wave >> 2;          // 0..1 -> 64 rows each
    const int wn   = wave & 3;           // 0..3 -> 32 cols each

    const int n0 = blockIdx.x * TN;      // E offset
    const int m0 = blockIdx.y * TM;      // patch-row offset

    // ---- per-thread global load coordinates (2 threads per tile row) ----
    const int rloc = tid >> 1;           // 0..127 : row within A/B tile
    const int seg  = tid & 1;            // which 16-wide half of the 32 K-cols

    // A-side: decode patch row m -> (img, ph, pw)
    const int m   = m0 + rloc;
    const int img = m / NPATCH;
    const int p   = m - img * NPATCH;
    const int ph  = p / 14;
    const int pw  = p - ph * 14;
    // running pointer, starts at tile ks=0: c=0, py=seg
    const float* aptr =
        x + (((size_t)img * 3) * 224 + (size_t)ph * 16 + seg) * 224 + pw * 16;
    // B-side: row e = n0 + rloc of W, seg selects 16 of the 32 K-cols
    const float* bptr = w + (size_t)(n0 + rloc) * GEMM_K + seg * 16;

    v8f acc[4][2] = {};                  // 4 M-frags x 2 N-frags per wave

    const int aRow = wm * 64;            // wave A-row base in tile
    const int bCol = wn * 32;            // wave B-col base in tile
    const int l16  = lane & 15;
    const int hi   = lane >> 4;          // 0 or 1 (half-wave)

    // prefetch registers (one full per-thread tile slice: 2 x 64B)
    v4f ra0, ra1, ra2, ra3, rb0, rb1, rb2, rb3;

    auto gload = [&]() {
        ra0 = *(const v4f*)(aptr + 0);  ra1 = *(const v4f*)(aptr + 4);
        ra2 = *(const v4f*)(aptr + 8);  ra3 = *(const v4f*)(aptr + 12);
        rb0 = *(const v4f*)(bptr + 0);  rb1 = *(const v4f*)(bptr + 4);
        rb2 = *(const v4f*)(bptr + 8);  rb3 = *(const v4f*)(bptr + 12);
    };

    // advance pointers past tile `ks` (the one just loaded)
    auto advance = [&](int ks) {
        aptr += ((ks & 7) == 7) ? A_STEP_CH : A_STEP_IN;
        bptr += KSTEP;
    };

    auto cvt_store = [&](int buf) {
        _Float16* ap = &As[buf][rloc * LDA + seg * 16];
        *(v8h*)(ap)     = cvt8(ra0, ra1);
        *(v8h*)(ap + 8) = cvt8(ra2, ra3);
        _Float16* bp = &Bs[buf][rloc * LDA + seg * 16];
        *(v8h*)(bp)     = cvt8(rb0, rb1);
        *(v8h*)(bp + 8) = cvt8(rb2, rb3);
    };

    // Load all 6 fragments into distinct live values: 12 ds_load_b128
    // in flight at once -> a single dscnt wait before the WMMA burst.
    auto frag_load = [&](int buf, v16h* af, v16h* bf) {
        #pragma unroll
        for (int nf = 0; nf < 2; ++nf) {
            // lane holds col (bCol + nf*16 + l16), K {0-15 | 16-31}
            const _Float16* bp = &Bs[buf][(bCol + nf * 16 + l16) * LDA + hi * 16];
            bf[nf] = ld16(bp, bp + 8);
        }
        #pragma unroll
        for (int mf = 0; mf < 4; ++mf) {
            // lane holds row, K {0-15 | 16-31}
            const _Float16* ap = &As[buf][(aRow + mf * 16 + l16) * LDA + hi * 16];
            af[mf] = ld16(ap, ap + 8);
        }
    };

    auto wmma_all = [&](const v16h* af, const v16h* bf) {
        #pragma unroll
        for (int mf = 0; mf < 4; ++mf)
            #pragma unroll
            for (int nf = 0; nf < 2; ++nf)
                acc[mf][nf] = __builtin_amdgcn_wmma_f32_16x16x32_f16(
                    false, af[mf], false, bf[nf],
                    (short)0, acc[mf][nf], false, false);
    };

    // ---- software pipeline: 1 barrier per K-step, double-buffered LDS ----
    gload();                // tile 0
    advance(0);
    cvt_store(0);
    gload();                // tile 1
    advance(1);
    __syncthreads();

    #pragma unroll 2
    for (int ks = 0; ks < NK - 2; ++ks) {
        v16h af[4], bf[2];
        frag_load(ks & 1, af, bf);   // 12 ds_loads (issued first)
        cvt_store((ks + 1) & 1);     // 4 ds_stores (drain under WMMAs)
        gload();                     // prefetch tile ks+2
        advance(ks + 2);
        wmma_all(af, bf);            // 8 WMMAs, wait dscnt<=4 once
        __syncthreads();
    }
    {   // tail: tiles NK-2 and NK-1
        v16h af[4], bf[2];
        frag_load((NK - 2) & 1, af, bf);
        cvt_store((NK - 1) & 1);
        wmma_all(af, bf);
        __syncthreads();
        frag_load((NK - 1) & 1, af, bf);
        wmma_all(af, bf);
    }

    // ---- epilogue: C/D layout -> row = r + 8*hi, col = l16 ----
    #pragma unroll
    for (int nf = 0; nf < 2; ++nf) {
        const int col = n0 + bCol + nf * 16 + l16;
        const float bv = bias[col];
        #pragma unroll
        for (int mf = 0; mf < 4; ++mf) {
            const int rbase = m0 + aRow + mf * 16 + hi * 8;
            #pragma unroll
            for (int r = 0; r < 8; ++r)
                out[(size_t)(rbase + r) * GEMM_N + col] = acc[mf][nf][r] + bv;
        }
    }
}

extern "C" void kernel_launch(void* const* d_in, const int* in_sizes, int n_in,
                              void* d_out, int out_size, void* d_ws, size_t ws_size,
                              hipStream_t stream) {
    (void)in_sizes; (void)n_in; (void)out_size; (void)d_ws; (void)ws_size;
    const float* x = (const float*)d_in[0];   // [128,3,224,224]
    const float* w = (const float*)d_in[1];   // [768,3,16,16] == [768][768]
    const float* b = (const float*)d_in[2];   // [768]
    float* out = (float*)d_out;               // [128,196,768]

    dim3 grid(GEMM_N / TN, (128 * NPATCH) / TM);  // (6, 196)
    patch_embed_wmma<<<grid, NTHREADS, 0, stream>>>(x, w, b, out);
}
